// CompactBilinearPooling_39556648796707
// MI455X (gfx1250) — matrix operand order repaired and checked by reading, not compiled, not run on hardware
//
#include <hip/hip_runtime.h>
#include <math.h>

#define B_  32
#define C_  512
#define N_  784    // 28*28
#define O_  1024

typedef __attribute__((ext_vector_type(16))) __bf16 v16bf;
typedef __attribute__((ext_vector_type(8)))  __bf16 v8bf;
typedef __attribute__((ext_vector_type(8)))  float  v8f;

__device__ __forceinline__ float2 cmulf(float2 a, float2 b) {
    return make_float2(a.x * b.x - a.y * b.y, a.x * b.y + a.y * b.x);
}

// Intra-wave LDS producer->consumer ordering: the wave is lockstep, but LDS
// ops retire via DScnt. Wait for our own DS ops; memory clobber stops the
// compiler from caching LDS values across the sync.
__device__ __forceinline__ void wave_lds_sync() {
    asm volatile("s_wait_dscnt 0" ::: "memory");
}

__device__ __forceinline__ int rev10(int x) {
    return (int)(__brev((unsigned)x) >> 22);
}

// ---------------------------------------------------------------------------
// Pack x [B, C, N] f32  ->  A [B, N, C] bf16 (tiled 32x32 transpose)
// ---------------------------------------------------------------------------
__global__ void pack_x_kernel(const float* __restrict__ x, __bf16* __restrict__ A) {
    __shared__ float tile[32][33];
    const int c0 = blockIdx.x * 32;
    const int n0 = blockIdx.y * 32;
    const int b  = blockIdx.z;
    const int t  = threadIdx.x;
#pragma unroll
    for (int i = 0; i < 4; ++i) {
        int idx = t + i * 256;
        int nl = idx & 31, cl = idx >> 5;
        int n = n0 + nl;
        float v = 0.0f;
        if (n < N_) v = x[((size_t)b * C_ + (c0 + cl)) * N_ + n];
        tile[cl][nl] = v;
    }
    __syncthreads();
#pragma unroll
    for (int i = 0; i < 4; ++i) {
        int idx = t + i * 256;
        int cl = idx & 31, nl = idx >> 5;
        int n = n0 + nl;
        if (n < N_)
            A[((size_t)b * N_ + n) * C_ + (c0 + cl)] = (__bf16)tile[cl][nl];
    }
}

// ---------------------------------------------------------------------------
// Pack sketch [O, C] f32 -> bf16 (same layout)
// ---------------------------------------------------------------------------
__global__ void pack_s_kernel(const float* __restrict__ s, __bf16* __restrict__ Sb, int n) {
    int i = blockIdx.x * 256 + threadIdx.x;
    if (i < n) Sb[i] = (__bf16)s[i];
}

// ---------------------------------------------------------------------------
// Fused dual-GEMM (WMMA bf16) + in-place FFT circular convolution.
// Block: 512 threads = 16 waves. Covers 16 n-rows x full O=1024 of one batch.
//   GEMM: wave w owns o-tiles [w*64 .. w*64+63], both projections.
//   FFT : wave w owns row w, fully in-place (DIF fwd -> pointwise -> DIT inv).
// LDS: zbuf 16x1024 float2 (128 KB) + twiddle 512 float2 (4 KB) = 132 KB
//      -> two blocks per WGP (8 waves/SIMD).
// ---------------------------------------------------------------------------
__launch_bounds__(512)
__global__ void cbp_fused_kernel(const __bf16* __restrict__ A1,
                                 const __bf16* __restrict__ A2,
                                 const __bf16* __restrict__ S1,
                                 const __bf16* __restrict__ S2,
                                 float* __restrict__ creal) {
    extern __shared__ float smem[];
    float2* zbuf = (float2*)smem;            // 16*1024 float2
    float2* tw   = zbuf + 16 * 1024;         // 512 float2: tw[j]=e^{-2pi i j/1024}

    const int tid  = threadIdx.x;
    const int lane = tid & 31;
    const int wave = tid >> 5;
    const int b     = blockIdx.y;
    const int ntile = blockIdx.x;            // 0..48

    if (tid < 512) {
        float ang = -2.0f * 3.14159265358979323846f * (float)tid * (1.0f / 1024.0f);
        float sv, cv;
        __sincosf(ang, &sv, &cv);
        tw[tid] = make_float2(cv, sv);
    }

    // ---------------- GEMM phase ----------------
    const int m  = lane & 15;                // M (n-row) / N (o-col) index in tile
    const int hi = lane >> 4;                // lane half
    const int kA = hi * 16;                  // A-frag K base (contiguous, matches B)
    const int kB = hi * 16;                  // B-frag K base (contiguous, matches A)
    const int nrow = ntile * 16 + m;

    const __bf16* a1p = A1 + ((size_t)b * N_ + nrow) * C_;
    const __bf16* a2p = A2 + ((size_t)b * N_ + nrow) * C_;

    v8f acc1[4], acc2[4];
#pragma unroll
    for (int i = 0; i < 4; ++i)
#pragma unroll
        for (int j = 0; j < 8; ++j) { acc1[i][j] = 0.0f; acc2[i][j] = 0.0f; }

    // Rolling B prefetch: keep next (ot,k) step's B fragments in flight while
    // the WMMAs consume the current ones.
    v16bf b1c, b2c;
    {
        int o0 = wave * 64 + m;
        b1c = *(const v16bf*)(S1 + (size_t)o0 * C_ + kB);
        b2c = *(const v16bf*)(S2 + (size_t)o0 * C_ + kB);
    }

#pragma unroll 1
    for (int k0 = 0; k0 < C_; k0 += 32) {
        v8bf a1lo = *(const v8bf*)(a1p + k0 + kA);
        v8bf a1hi = *(const v8bf*)(a1p + k0 + 8 + kA);
        v8bf a2lo = *(const v8bf*)(a2p + k0 + kA);
        v8bf a2hi = *(const v8bf*)(a2p + k0 + 8 + kA);
        v16bf a1, a2;
#pragma unroll
        for (int i = 0; i < 8; ++i) {
            a1[i] = a1lo[i]; a1[i + 8] = a1hi[i];
            a2[i] = a2lo[i]; a2[i + 8] = a2hi[i];
        }
#pragma unroll
        for (int ot = 0; ot < 4; ++ot) {
            v16bf b1n = b1c, b2n = b2c;
            const int ot_n = (ot + 1) & 3;
            const int k_n  = (ot == 3) ? (k0 + 32) : k0;
            if (k_n < C_) {
                int o_n = wave * 64 + ot_n * 16 + m;
                b1n = *(const v16bf*)(S1 + (size_t)o_n * C_ + k_n + kB);
                b2n = *(const v16bf*)(S2 + (size_t)o_n * C_ + k_n + kB);
            }
            acc1[ot] = __builtin_amdgcn_wmma_f32_16x16x32_bf16(
                false, a1, false, b1c, (short)0, acc1[ot], false, false);
            acc2[ot] = __builtin_amdgcn_wmma_f32_16x16x32_bf16(
                false, a2, false, b2c, (short)0, acc2[ot], false, false);
            b1c = b1n; b2c = b2n;
        }
    }

    // Deposit packed complex z = p1 + i*p2 into LDS.
    // C layout: VGPR r -> M = r (lanes 0-15) / r+8 (lanes 16-31); col = m.
#pragma unroll
    for (int ot = 0; ot < 4; ++ot) {
        int o = wave * 64 + ot * 16 + m;
#pragma unroll
        for (int r = 0; r < 8; ++r) {
            int nl = r + hi * 8;
            zbuf[nl * 1024 + o] = make_float2(acc1[ot][r], acc2[ot][r]);
        }
    }
    __syncthreads();

    // ---------------- FFT-conv phase (wave owns row `wave`, in-place) -------
    float2* row = zbuf + wave * 1024;

    // Forward FFT, DIF: natural in -> bit-reversed out. tw index k*2^(9-lg).
#pragma unroll 1
    for (int lg = 9; lg >= 0; --lg) {
        const int Ns = 1 << lg;
#pragma unroll 4
        for (int t2 = lane; t2 < 512; t2 += 32) {
            int k = t2 & (Ns - 1);
            int i = ((t2 >> lg) << (lg + 1)) + k;
            float2 a = row[i];
            float2 bv = row[i + Ns];
            float2 w = tw[k << (9 - lg)];
            row[i] = make_float2(a.x + bv.x, a.y + bv.y);
            float2 d = make_float2(a.x - bv.x, a.y - bv.y);
            row[i + Ns] = cmulf(d, w);
        }
        wave_lds_sync();
    }

    // Pointwise in bit-reversed domain. Position p holds Z(rev10(p)).
    // P(k)=(Z(k)+conj(Z(N-k)))/2, Q(k)=(Z(k)-conj(Z(N-k)))/(2i), C=P*Q,
    // C(N-k)=conj(C(k)). Pairs {k, N-k} are disjoint; k=0,512 self-paired.
    for (int k = lane; k <= 512; k += 32) {
        int p = rev10(k);
        int q = rev10((1024 - k) & 1023);
        float2 Zk = row[p];
        float2 Zr = row[q];
        float2 P = make_float2(0.5f * (Zk.x + Zr.x), 0.5f * (Zk.y - Zr.y));
        float2 D = make_float2(Zk.x - Zr.x, Zk.y + Zr.y);   // Zk - conj(Zr)
        float2 Q = make_float2(0.5f * D.y, -0.5f * D.x);    // -0.5i * D
        float2 Cc = cmulf(P, Q);
        row[p] = Cc;
        row[q] = make_float2(Cc.x, -Cc.y);
    }
    wave_lds_sync();

    // Inverse FFT, DIT: bit-reversed in -> natural out; conjugated twiddles.
#pragma unroll 1
    for (int lg = 0; lg <= 9; ++lg) {
        const int Ns = 1 << lg;
#pragma unroll 4
        for (int t2 = lane; t2 < 512; t2 += 32) {
            int k = t2 & (Ns - 1);
            int i = ((t2 >> lg) << (lg + 1)) + k;
            float2 w = tw[k << (9 - lg)];
            float2 a = row[i];
            float2 bw = cmulf(row[i + Ns], make_float2(w.x, -w.y));
            row[i]      = make_float2(a.x + bw.x, a.y + bw.y);
            row[i + Ns] = make_float2(a.x - bw.x, a.y - bw.y);
        }
        wave_lds_sync();
    }

    // Real part, scaled by 1/N, coalesced per wave.
    float* dst = creal + ((size_t)b * N_ + (size_t)ntile * 16 + wave) * O_;
#pragma unroll 4
    for (int k = lane; k < 1024; k += 32) {
        dst[k] = row[k].x * (1.0f / 1024.0f);
    }
}

// ---------------------------------------------------------------------------
// creal [B, N, O] -> out [B, O, N] (tiled 32x32 transpose)
// ---------------------------------------------------------------------------
__global__ void transpose_out_kernel(const float* __restrict__ creal,
                                     float* __restrict__ out) {
    __shared__ float tile[32][33];
    const int o0 = blockIdx.x * 32;
    const int n0 = blockIdx.y * 32;
    const int b  = blockIdx.z;
    const int t  = threadIdx.x;
#pragma unroll
    for (int i = 0; i < 4; ++i) {
        int idx = t + i * 256;
        int ol = idx & 31, nl = idx >> 5;
        int n = n0 + nl;
        float v = 0.0f;
        if (n < N_) v = creal[((size_t)b * N_ + n) * O_ + (o0 + ol)];
        tile[nl][ol] = v;
    }
    __syncthreads();
#pragma unroll
    for (int i = 0; i < 4; ++i) {
        int idx = t + i * 256;
        int nl = idx & 31, ol = idx >> 5;
        int n = n0 + nl;
        if (n < N_)
            out[((size_t)b * O_ + (o0 + ol)) * N_ + n] = tile[nl][ol];
    }
}

// ---------------------------------------------------------------------------
extern "C" void kernel_launch(void* const* d_in, const int* in_sizes, int n_in,
                              void* d_out, int out_size, void* d_ws, size_t ws_size,
                              hipStream_t stream) {
    const float* x1 = (const float*)d_in[0];
    const float* x2 = (const float*)d_in[1];
    const float* s1 = (const float*)d_in[2];
    const float* s2 = (const float*)d_in[3];
    float* out = (float*)d_out;

    char* ws = (char*)d_ws;
    const size_t szA = (size_t)B_ * N_ * C_ * sizeof(__bf16);   // 25,690,112 B
    const size_t szS = (size_t)O_ * C_ * sizeof(__bf16);        //  1,048,576 B
    __bf16* A1 = (__bf16*)(ws);
    __bf16* A2 = (__bf16*)(ws + szA);
    __bf16* S1 = (__bf16*)(ws + 2 * szA);
    __bf16* S2 = (__bf16*)(ws + 2 * szA + szS);
    float* creal = (float*)(ws + 2 * szA + 2 * szS);            // B*N*O f32

    // Pack inputs to bf16 (A transposed to [B, N, C])
    dim3 gx(C_ / 32, (N_ + 31) / 32, B_);
    pack_x_kernel<<<gx, 256, 0, stream>>>(x1, A1);
    pack_x_kernel<<<gx, 256, 0, stream>>>(x2, A2);
    const int nS = O_ * C_;
    pack_s_kernel<<<(nS + 255) / 256, 256, 0, stream>>>(s1, S1, nS);
    pack_s_kernel<<<(nS + 255) / 256, 256, 0, stream>>>(s2, S2, nS);

    // Fused WMMA GEMM + in-place FFT circular convolution
    const size_t ldsBytes = (size_t)(16 * 1024 + 512) * sizeof(float2); // 135,168 B
    (void)hipFuncSetAttribute((const void*)cbp_fused_kernel,
                              hipFuncAttributeMaxDynamicSharedMemorySize,
                              (int)ldsBytes);
    cbp_fused_kernel<<<dim3(N_ / 16, B_), 512, ldsBytes, stream>>>(
        A1, A2, S1, S2, creal);

    // Final transpose to [B, O, H, W]
    transpose_out_kernel<<<dim3(O_ / 32, (N_ + 31) / 32, B_), 256, 0, stream>>>(
        creal, out);
}
